// MixLinear_FP8GEMM_5600637354124
// MI455X (gfx1250) — 8-bit matrix operand order repaired and checked by reading, compiled, not run on hardware
//
#include <hip/hip_runtime.h>

#define GEMM_M 16384
#define GEMM_N 4096
#define GEMM_K 4096
#define FRAG_BYTES 2048
#define KBLOCKS (GEMM_K / 128)

typedef __attribute__((ext_vector_type(16))) int   v16i;
typedef __attribute__((ext_vector_type(8)))  float v8f;

__device__ __forceinline__ unsigned int h_to_e5m2(unsigned int h) {
  unsigned int sign = (h >> 8) & 0x80u;
  unsigned int a = h & 0x7fffu;
  if (a > 0x7c00u) return sign | 0x7fu;
  a += 0x7fu + ((a >> 8) & 1u);
  if (a > 0x7c00u) a = 0x7c00u;
  return sign | (a >> 8);
}

__device__ __forceinline__ unsigned int f_to_e4m3(float f) {
  unsigned int u = __float_as_uint(f);
  unsigned int sign = (u >> 24) & 0x80u;
  unsigned int a = u & 0x7fffffffu;
  if (a >= 0x7f800000u) return sign | 0x7fu;
  if (a >= 0x43e80000u) return sign | 0x7eu;
  if (a < 0x3c800000u) {
    float m = rintf(__uint_as_float(a) * 512.0f);
    return sign | (unsigned int)m;
  }
  unsigned int e    = (a >> 23) - 127u + 7u;
  unsigned int mant = a & 0x7fffffu;
  unsigned int keep = mant >> 20;
  unsigned int rem  = mant & 0xfffffu;
  keep += (rem > 0x80000u) || (rem == 0x80000u && (keep & 1u));
  if (keep == 8u) { keep = 0u; e += 1u; }
  unsigned int code = (e << 3) | keep;
  if (code > 0x7eu) code = 0x7eu;
  return sign | code;
}

__device__ __forceinline__ float half_bits_to_float(unsigned int h) {
  unsigned short s = (unsigned short)h;
  _Float16 v;
  __builtin_memcpy(&v, &s, 2);
  return (float)v;
}

__global__ __launch_bounds__(256) void quant_pack_a_kernel(
    const unsigned short* __restrict__ x, unsigned char* __restrict__ dst) {
  const int i    = blockIdx.x * blockDim.x + threadIdx.x;
  const int frag = i >> 8;
  const int r    = i & 255;
  const int lane = r & 31;
  const int c    = r >> 5;
  const int ll   = lane & 15;
  const int hi   = lane >> 4;
  const int tileM = frag >> 5;
  const int tileK = frag & (KBLOCKS - 1);

  const int m = tileM * 16 + ll;
  const int k = tileK * 128 + ((c >> 1) * 32) + hi * 16 + ((c & 1) * 8);

  const uint4 v = *(const uint4*)(x + (size_t)m * GEMM_K + k);
  unsigned int b0 =  h_to_e5m2(v.x & 0xffffu)
                  | (h_to_e5m2(v.x >> 16)     << 8)
                  | (h_to_e5m2(v.y & 0xffffu) << 16)
                  | (h_to_e5m2(v.y >> 16)     << 24);
  unsigned int b1 =  h_to_e5m2(v.z & 0xffffu)
                  | (h_to_e5m2(v.z >> 16)     << 8)
                  | (h_to_e5m2(v.w & 0xffffu) << 16)
                  | (h_to_e5m2(v.w >> 16)     << 24);

  const size_t off = (size_t)frag * FRAG_BYTES + (c >> 1) * 512 + lane * 16 + (c & 1) * 8;
  *(uint2*)(dst + off) = make_uint2(b0, b1);
}

__global__ __launch_bounds__(256) void quant_pack_b_kernel(
    const unsigned short* __restrict__ w, unsigned char* __restrict__ dst) {
  const int i    = blockIdx.x * blockDim.x + threadIdx.x;
  const int frag = i >> 7;
  const int r    = i & 127;
  const int lane = r & 31;
  const int q    = r >> 5;
  const int ll   = lane & 15;
  const int hi   = lane >> 4;
  const int tileN = frag >> 5;
  const int tileK = frag & (KBLOCKS - 1);

  const int n = tileN * 16 + ll;
  const int k = tileK * 128 + q * 32 + hi * 16;

  const unsigned short* src = w + (size_t)n * GEMM_K + k;
  const uint4 v0 = *(const uint4*)(src);
  const uint4 v1 = *(const uint4*)(src + 8);

  uint4 o;
  o.x =  f_to_e4m3(half_bits_to_float(v0.x & 0xffffu))
      | (f_to_e4m3(half_bits_to_float(v0.x >> 16))     << 8)
      | (f_to_e4m3(half_bits_to_float(v0.y & 0xffffu)) << 16)
      | (f_to_e4m3(half_bits_to_float(v0.y >> 16))     << 24);
  o.y =  f_to_e4m3(half_bits_to_float(v0.z & 0xffffu))
      | (f_to_e4m3(half_bits_to_float(v0.z >> 16))     << 8)
      | (f_to_e4m3(half_bits_to_float(v0.w & 0xffffu)) << 16)
      | (f_to_e4m3(half_bits_to_float(v0.w >> 16))     << 24);
  o.z =  f_to_e4m3(half_bits_to_float(v1.x & 0xffffu))
      | (f_to_e4m3(half_bits_to_float(v1.x >> 16))     << 8)
      | (f_to_e4m3(half_bits_to_float(v1.y & 0xffffu)) << 16)
      | (f_to_e4m3(half_bits_to_float(v1.y >> 16))     << 24);
  o.w =  f_to_e4m3(half_bits_to_float(v1.z & 0xffffu))
      | (f_to_e4m3(half_bits_to_float(v1.z >> 16))     << 8)
      | (f_to_e4m3(half_bits_to_float(v1.w & 0xffffu)) << 16)
      | (f_to_e4m3(half_bits_to_float(v1.w >> 16))     << 24);

  const size_t off = (size_t)frag * FRAG_BYTES + q * 512 + lane * 16;
  *(uint4*)(dst + off) = o;
}

__global__ __launch_bounds__(256) void fp8gemm_kernel(
    const unsigned char* __restrict__ A8,
    const unsigned char* __restrict__ W8,
    const _Float16*      __restrict__ bias,
    _Float16*            __restrict__ out)
{
  const int lane = threadIdx.x & 31;
  const int wave = threadIdx.x >> 5;
  const int ll   = lane & 15;
  const int hi   = lane >> 4;
  const int wm   = wave & 1;
  const int wn   = wave >> 1;

  const int m0 = blockIdx.y * 64  + wm * 32;
  const int n0 = blockIdx.x * 256 + wn * 64;

  v8f acc[2][4] = {};

  const size_t MROW = (size_t)KBLOCKS * FRAG_BYTES;
  const unsigned char* pa = A8 + (size_t)(m0 >> 4) * MROW + lane * 16;
  const unsigned char* pb = W8 + (size_t)(n0 >> 4) * MROW + lane * 16;

  for (int t = 0; t < KBLOCKS; ++t) {
    v16i a[2];
    v16i b[4];
    const unsigned char* pat = pa + (size_t)t * FRAG_BYTES;
#pragma unroll
    for (int j = 0; j < 4; ++j) {
      const int4 q0 = *(const int4*)(pat + j * 512);
      const int4 q1 = *(const int4*)(pat + MROW + j * 512);
      a[0][4 * j]     = q0.x;  a[0][4 * j + 1] = q0.y;
      a[0][4 * j + 2] = q0.z;  a[0][4 * j + 3] = q0.w;
      a[1][4 * j]     = q1.x;  a[1][4 * j + 1] = q1.y;
      a[1][4 * j + 2] = q1.z;  a[1][4 * j + 3] = q1.w;
    }
#pragma unroll
    for (int nt = 0; nt < 4; ++nt) {
      const unsigned char* pbt = pb + (size_t)nt * MROW + (size_t)t * FRAG_BYTES;
#pragma unroll
      for (int j = 0; j < 4; ++j) {
        const int4 v = *(const int4*)(pbt + j * 512);
        b[nt][4 * j]     = v.x;  b[nt][4 * j + 1] = v.y;
        b[nt][4 * j + 2] = v.z;  b[nt][4 * j + 3] = v.w;
      }
    }
#pragma unroll
    for (int mt = 0; mt < 2; ++mt)
#pragma unroll
      for (int nt = 0; nt < 4; ++nt)
        acc[mt][nt] = __builtin_amdgcn_wmma_f32_16x16x128_bf8_fp8(
            a[mt], b[nt], (short)0, acc[mt][nt], false, false);
  }

#pragma unroll
  for (int nt = 0; nt < 4; ++nt) {
    const int n = n0 + nt * 16 + ll;
    const float bv = (float)bias[n];
#pragma unroll
    for (int mt = 0; mt < 2; ++mt) {
      const int mb = m0 + mt * 16 + hi * 8;
#pragma unroll
      for (int r = 0; r < 8; ++r)
        out[(size_t)(mb + r) * GEMM_N + n] = (_Float16)(acc[mt][nt][r] + bv);
    }
  }
}

extern "C" void kernel_launch(void* const* d_in, const int* in_sizes, int n_in,
                              void* d_out, int out_size, void* d_ws, size_t ws_size,
                              hipStream_t stream) {
  const unsigned short* x = (const unsigned short*)d_in[0];
  const unsigned short* w = (const unsigned short*)d_in[1];
  const _Float16* bias = (const _Float16*)d_in[2];
  _Float16* out = (_Float16*)d_out;

  unsigned char* A8 = (unsigned char*)d_ws;
  unsigned char* W8 = A8 + (size_t)GEMM_M * GEMM_K;

  const int na = GEMM_M * GEMM_K / 8;
  const int nb = GEMM_N * GEMM_K / 16;
  quant_pack_a_kernel<<<na / 256, 256, 0, stream>>>(x, A8);
  quant_pack_b_kernel<<<nb / 256, 256, 0, stream>>>(w, W8);

  dim3 grid(GEMM_N / 256, GEMM_M / 64);
  fp8gemm_kernel<<<grid, 256, 0, stream>>>(A8, W8, bias, out);
}
